// EdgeSAGE_14886356648674
// MI455X (gfx1250) — matrix operand order repaired and checked by reading, not compiled, not run on hardware
//
#include <hip/hip_runtime.h>
#include <stddef.h>
#include <stdint.h>

// ---------------------------------------------------------------------------
// EdgeSAGE GNN forward for MI455X (gfx1250, wave32, WMMA).
// N=50000 nodes, E=800000 edges, H=128, L=3 layers. All heavy GEMMs run on
// v_wmma_f32_16x16x32_bf16 (bf16 operands, f32 accum). Edge MLP fused with
// the segment-sum scatter (global_atomic_add_f32); h stays L2-resident.
//
// Round-1: opaque asm barrier on weight pointers inside the tile loop stops
//          LICM from hoisting 96 B-fragment loads (was spilling to scratch).
// Round-2: the barrier now returns an addrspace(1) pointer so B-fragment
//          loads lower to global_load_b128 (LOADcnt only) instead of
//          flat_load_b128 (LOADcnt+DScnt, ties up the LDS path).
// ---------------------------------------------------------------------------

typedef __attribute__((ext_vector_type(16))) __bf16 v16bf;
typedef __attribute__((ext_vector_type(8)))  float  v8f;

#define AS1 __attribute__((address_space(1)))

#define EW 8   // waves per block, edge kernel
#define NW 8   // waves per block, node kernel

// ---- helpers --------------------------------------------------------------

__device__ __forceinline__ int bcast_i(int v, int srcLane) {
  return __builtin_amdgcn_ds_bpermute(srcLane << 2, v);
}
__device__ __forceinline__ float bcast_f(float v, int srcLane) {
  return __int_as_float(__builtin_amdgcn_ds_bpermute(srcLane << 2, __float_as_int(v)));
}
__device__ __forceinline__ unsigned int pk2(float a, float b) {
  union { __bf16 h[2]; unsigned int u; } t;
  t.h[0] = (__bf16)a; t.h[1] = (__bf16)b;
  return t.u;
}

// Opaque pointer barrier: defeats LICM (so weight-fragment loads are re-issued
// per tile and never spilled) and returns a global-addrspace pointer so the
// loads lower to global_load_b128 rather than flat_load_b128.
__device__ __forceinline__ const AS1 __bf16* opaque_g(const __bf16* p) {
  unsigned long long u = (unsigned long long)(uintptr_t)p;
  asm volatile("" : "+v"(u));
  return (const AS1 __bf16*)(uintptr_t)u;
}

// A-fragment (16x32 bf16, ISA 7.12.2 layout) gathered from an LDS tile of
// row stride `stride` bf16 elements, starting at column `kbase`.
__device__ __forceinline__ v16bf ld_afrag(const __bf16* base, int lane, int kbase, int stride) {
  const int half = lane >> 4;
  const int row  = lane & 15;
  union { v16bf v; unsigned int u[8]; } f;
  const __bf16* rp = base + row * stride + kbase + half * 8;
#pragma unroll
  for (int p = 0; p < 8; ++p) {
    const int group  = p >> 2;   // K region 0-15 vs 16-31
    const int within = p & 3;    // pair within region
    f.u[p] = *(const unsigned int*)(rp + group * 16 + within * 2);
  }
  return f.v;
}

// B-fragment (32x16 bf16) from weights pre-packed in fragment order:
// 16 contiguous bf16 per lane, 32 lanes per fragment. Global addrspace ->
// two global_load_b128 per lane.
__device__ __forceinline__ v16bf ld_bfrag(const AS1 __bf16* packed, int frag, int lane) {
  return *(const AS1 v16bf*)(packed + (((size_t)frag << 5) + lane) * 16);
}

// ---- small utility kernels ------------------------------------------------

__global__ void zero_f32_k(float* __restrict__ p, long n) {
  for (long i = (long)blockIdx.x * blockDim.x + threadIdx.x; i < n; i += (long)gridDim.x * blockDim.x)
    p[i] = 0.0f;
}

__global__ void deg_k(const int* __restrict__ dst, float* __restrict__ deg, int E) {
  int i = blockIdx.x * blockDim.x + threadIdx.x;
  if (i < E) atomicAdd(deg + dst[i], 1.0f);
}

__global__ void deginv_k(const float* __restrict__ deg, float* __restrict__ deg_inv, int N) {
  int i = blockIdx.x * blockDim.x + threadIdx.x;
  if (i < N) deg_inv[i] = 1.0f / fmaxf(deg[i], 1.0f);
}

__global__ void drone_embed_k(const float* __restrict__ drone, const float* __restrict__ Wd,
                              const float* __restrict__ bd, float* __restrict__ demb, int B) {
  int t = blockIdx.x * blockDim.x + threadIdx.x;
  if (t >= B * 128) return;
  int b = t >> 7, c = t & 127;
  float acc = bd[c];
  for (int k = 0; k < 51; ++k) acc += drone[b * 51 + k] * Wd[k * 128 + c];
  demb[t] = acc;
}

__global__ void node_embed_k(const float* __restrict__ x, const float* __restrict__ Wn,
                             const float* __restrict__ bn, const float* __restrict__ demb,
                             const int* __restrict__ batch, float* __restrict__ h, int N) {
  long t = (long)blockIdx.x * blockDim.x + threadIdx.x;
  if (t >= (long)N * 128) return;
  int n = (int)(t >> 7), c = (int)(t & 127);
  float acc = bn[c] + demb[batch[n] * 128 + c];
#pragma unroll
  for (int k = 0; k < 16; ++k) acc += x[n * 16 + k] * Wn[k * 128 + c];
  h[t] = acc;
}

// Repack KxN f32 weight (N==128) into bf16 WMMA B-fragment order.
__global__ void pack_w_k(const float* __restrict__ src, __bf16* __restrict__ dst, int K) {
  int t = blockIdx.x * blockDim.x + threadIdx.x;
  if (t >= K * 128) return;
  int k = t >> 7, n = t & 127;
  int kc = k >> 5, rem = k & 31;
  int half = (rem >> 3) & 1, r2 = (rem & 7) | ((rem >> 4) << 3);
  int nt = n >> 4;
  int lane = half * 16 + (n & 15);
  size_t idx = ((((size_t)kc * 8 + nt) * 32 + lane) << 4) + r2;  // 16 bf16 per lane
  dst[idx] = (__bf16)src[(size_t)k * 128 + n];
}

// ---- edge MLP + scatter (the WMMA workhorse) ------------------------------
// One wave per tile of 16 edges:
//   m  = [x_j, x_j - x_i]                  (16x256 bf16 in LDS)
//   t  = relu(m @ We1 + be1)               (8kc x 8nt WMMAs, K=256)
//   mm = t @ We2 + be2                     (4kc x 8nt WMMAs, K=128)
//   aggr[dst] += mm                        (global_atomic_add_f32)
__global__ void __launch_bounds__(EW * 32)
edge_mlp_scatter_k(const float* __restrict__ h,
                   const int* __restrict__ src, const int* __restrict__ dst,
                   const __bf16* __restrict__ pWe1, const __bf16* __restrict__ pWe2,
                   const float* __restrict__ be1, const float* __restrict__ be2,
                   float* __restrict__ aggr, int numTiles) {
  __shared__ __attribute__((aligned(16))) __bf16 mlds[EW][16][256];  // 64 KB
  const int wave = threadIdx.x >> 5;
  const int lane = threadIdx.x & 31;
  const int half = lane >> 4;
  const int c0   = lane * 4;            // 4 f32 columns per lane
  __bf16 (*m)[256] = mlds[wave];

  // keep the (cache-resident) packed weights warm
  __builtin_prefetch(pWe1 + (size_t)threadIdx.x * 128, 0, 1);
  __builtin_prefetch(pWe2 + (size_t)threadIdx.x * 64, 0, 1);

  for (int t = blockIdx.x * EW + wave; t < numTiles; t += gridDim.x * EW) {
    // per-iteration opaque weight pointers: loads cannot be hoisted/spilled
    const AS1 __bf16* w1 = opaque_g(pWe1);
    const AS1 __bf16* w2 = opaque_g(pWe2);

    const int e0 = t << 4;
    int sIdx = 0, dIdx = 0;
    if (lane < 16) { sIdx = src[e0 + lane]; dIdx = dst[e0 + lane]; }

    // --- stage m = [x_j, x_j - x_i] (gather through L2, convert to bf16) ---
#pragma unroll 1
    for (int i = 0; i < 16; ++i) {
      const int s = bcast_i(sIdx, i);
      const int d = bcast_i(dIdx, i);
      const float4 xj = *(const float4*)(h + ((size_t)s << 7) + c0);
      const float4 xi = *(const float4*)(h + ((size_t)d << 7) + c0);
      *(unsigned int*)&m[i][c0]           = pk2(xj.x, xj.y);
      *(unsigned int*)&m[i][c0 + 2]       = pk2(xj.z, xj.w);
      *(unsigned int*)&m[i][128 + c0]     = pk2(xj.x - xi.x, xj.y - xi.y);
      *(unsigned int*)&m[i][128 + c0 + 2] = pk2(xj.z - xi.z, xj.w - xi.w);
    }

    // --- GEMM1: (16x256) @ (256x128) -> acc -------------------------------
    v8f acc[8] = {};
#pragma unroll
    for (int kc = 0; kc < 8; ++kc) {
      const v16bf a = ld_afrag(&m[0][0], lane, kc * 32, 256);
#pragma unroll
      for (int nt = 0; nt < 8; ++nt) {
        const v16bf b = ld_bfrag(w1, kc * 8 + nt, lane);
        acc[nt] = __builtin_amdgcn_wmma_f32_16x16x32_bf16(
            false, a, false, b, (short)0, acc[nt], false, false);
      }
    }

    // --- relu + bias, write t back into LDS cols 0..127 -------------------
#pragma unroll
    for (int nt = 0; nt < 8; ++nt) {
      const int n = (lane & 15) + (nt << 4);
      const float bias = be1[n];
#pragma unroll
      for (int r = 0; r < 8; ++r) {
        const float v = acc[nt][r] + bias;
        m[r + 8 * half][n] = (__bf16)(v > 0.0f ? v : 0.0f);
      }
    }

    // --- GEMM2: (16x128) @ (128x128) -> acc2 ------------------------------
    v8f acc2[8] = {};
#pragma unroll
    for (int kc = 0; kc < 4; ++kc) {
      const v16bf a = ld_afrag(&m[0][0], lane, kc * 32, 256);
#pragma unroll
      for (int nt = 0; nt < 8; ++nt) {
        const v16bf b = ld_bfrag(w2, kc * 8 + nt, lane);
        acc2[nt] = __builtin_amdgcn_wmma_f32_16x16x32_bf16(
            false, a, false, b, (short)0, acc2[nt], false, false);
      }
    }

    // --- segment-sum scatter (bias added here) ----------------------------
#pragma unroll
    for (int nt = 0; nt < 8; ++nt) {
      const int n = (lane & 15) + (nt << 4);
      const float bias = be2[n];
#pragma unroll
      for (int r = 0; r < 8; ++r) {
        const int row = r + 8 * half;                 // edge within tile
        const int d = bcast_i(dIdx, row);
        atomicAdd(aggr + ((size_t)d << 7) + n, acc2[nt][r] + bias);
      }
    }
  }
}

// ---- per-layer node update ------------------------------------------------
// One wave per tile of 16 nodes:
//   hs = h @ Wself + bself             (4x8 WMMAs)
//   c  = [hs, aggr*deg_inv] @ Wout + b (8x8 WMMAs)
//   h += relu(layernorm(c))
__global__ void __launch_bounds__(NW * 32)
node_update_k(float* __restrict__ h, const float* __restrict__ aggr,
              const float* __restrict__ deg_inv,
              const __bf16* __restrict__ pWself, const __bf16* __restrict__ pWout,
              const float* __restrict__ bself, const float* __restrict__ bout,
              const float* __restrict__ lng, const float* __restrict__ lnb,
              int numTiles) {
  __shared__ __attribute__((aligned(16))) char smemraw[NW][16 * 256 * 2];  // 64 KB
  const int wave = threadIdx.x >> 5;
  const int lane = threadIdx.x & 31;
  const int half = lane >> 4;
  const int c0   = lane * 4;
  __bf16 (*A)[256] = (__bf16 (*)[256])smemraw[wave];   // bf16 view (concat input)
  float  (*C)[128] = (float  (*)[128])smemraw[wave];   // f32 view (post-GEMM c)

  const float4 gg  = *(const float4*)(lng + c0);
  const float4 bbv = *(const float4*)(lnb + c0);

  for (int t = blockIdx.x * NW + wave; t < numTiles; t += gridDim.x * NW) {
    const AS1 __bf16* ws_ = opaque_g(pWself);
    const AS1 __bf16* wo_ = opaque_g(pWout);
    const int n0 = t << 4;

    // stage h rows (bf16) into cols 0..127
#pragma unroll 1
    for (int i = 0; i < 16; ++i) {
      const float4 hv = *(const float4*)(h + ((size_t)(n0 + i) << 7) + c0);
      *(unsigned int*)&A[i][c0]     = pk2(hv.x, hv.y);
      *(unsigned int*)&A[i][c0 + 2] = pk2(hv.z, hv.w);
    }

    // GEMM1: hs = h @ Wself
    v8f accS[8] = {};
#pragma unroll
    for (int kc = 0; kc < 4; ++kc) {
      const v16bf a = ld_afrag(&A[0][0], lane, kc * 32, 256);
#pragma unroll
      for (int nt = 0; nt < 8; ++nt) {
        const v16bf b = ld_bfrag(ws_, kc * 8 + nt, lane);
        accS[nt] = __builtin_amdgcn_wmma_f32_16x16x32_bf16(
            false, a, false, b, (short)0, accS[nt], false, false);
      }
    }
    // write hs (+bias) over cols 0..127
#pragma unroll
    for (int nt = 0; nt < 8; ++nt) {
      const int n = (lane & 15) + (nt << 4);
      const float bias = bself[n];
#pragma unroll
      for (int r = 0; r < 8; ++r)
        A[r + 8 * half][n] = (__bf16)(accS[nt][r] + bias);
    }
    // stage aggr * deg_inv into cols 128..255
#pragma unroll 1
    for (int i = 0; i < 16; ++i) {
      const float di = deg_inv[n0 + i];
      const float4 av = *(const float4*)(aggr + ((size_t)(n0 + i) << 7) + c0);
      *(unsigned int*)&A[i][128 + c0]     = pk2(av.x * di, av.y * di);
      *(unsigned int*)&A[i][128 + c0 + 2] = pk2(av.z * di, av.w * di);
    }

    // GEMM2: c = [hs, aggr'] @ Wout
    v8f accO[8] = {};
#pragma unroll
    for (int kc = 0; kc < 8; ++kc) {
      const v16bf a = ld_afrag(&A[0][0], lane, kc * 32, 256);
#pragma unroll
      for (int nt = 0; nt < 8; ++nt) {
        const v16bf b = ld_bfrag(wo_, kc * 8 + nt, lane);
        accO[nt] = __builtin_amdgcn_wmma_f32_16x16x32_bf16(
            false, a, false, b, (short)0, accO[nt], false, false);
      }
    }
    // c (+bias) -> f32 LDS (overlays A; all A reads are complete)
#pragma unroll
    for (int nt = 0; nt < 8; ++nt) {
      const int n = (lane & 15) + (nt << 4);
      const float bias = bout[n];
#pragma unroll
      for (int r = 0; r < 8; ++r)
        C[r + 8 * half][n] = accO[nt][r] + bias;
    }

    // layernorm stats: lanes 0..15 own one row each
    float mu = 0.0f, rs = 0.0f;
    if (lane < 16) {
      float s = 0.0f, sq = 0.0f;
#pragma unroll 1
      for (int j = 0; j < 128; j += 4) {
        const float4 cv = *(const float4*)&C[lane][j];
        s  += cv.x + cv.y + cv.z + cv.w;
        sq += cv.x * cv.x + cv.y * cv.y + cv.z * cv.z + cv.w * cv.w;
      }
      mu = s * (1.0f / 128.0f);
      const float var = sq * (1.0f / 128.0f) - mu * mu;
      rs = rsqrtf(var + 1e-5f);
    }

    // h += relu(ln(c))
#pragma unroll 1
    for (int i = 0; i < 16; ++i) {
      const float mui = bcast_f(mu, i);
      const float rsi = bcast_f(rs, i);
      const float4 cv = *(const float4*)&C[i][c0];
      float4 hv = *(const float4*)(h + ((size_t)(n0 + i) << 7) + c0);
      float v;
      v = (cv.x - mui) * rsi * gg.x + bbv.x; hv.x += (v > 0.0f ? v : 0.0f);
      v = (cv.y - mui) * rsi * gg.y + bbv.y; hv.y += (v > 0.0f ? v : 0.0f);
      v = (cv.z - mui) * rsi * gg.z + bbv.z; hv.z += (v > 0.0f ? v : 0.0f);
      v = (cv.w - mui) * rsi * gg.w + bbv.w; hv.w += (v > 0.0f ? v : 0.0f);
      *(float4*)(h + ((size_t)(n0 + i) << 7) + c0) = hv;
    }
  }
}

// ---- final projection 128 -> 2 --------------------------------------------
__global__ void proj_k(const float* __restrict__ h, const float* __restrict__ Wp,
                       const float* __restrict__ bp, float* __restrict__ out, int N) {
  int n = blockIdx.x * blockDim.x + threadIdx.x;
  if (n >= N) return;
  float a0 = bp[0], a1 = bp[1];
#pragma unroll 4
  for (int k = 0; k < 128; ++k) {
    const float hv = h[((size_t)n << 7) + k];
    a0 += hv * Wp[2 * k];
    a1 += hv * Wp[2 * k + 1];
  }
  out[2 * n]     = a0;
  out[2 * n + 1] = a1;
}

// ---------------------------------------------------------------------------
extern "C" void kernel_launch(void* const* d_in, const int* in_sizes, int n_in,
                              void* d_out, int out_size, void* d_ws, size_t ws_size,
                              hipStream_t stream) {
  const float* x      = (const float*)d_in[0];
  const int*   eidx   = (const int*)  d_in[1];
  const float* drone  = (const float*)d_in[2];
  const int*   batch  = (const int*)  d_in[3];
  const float* W_node = (const float*)d_in[4];
  const float* b_node = (const float*)d_in[5];
  const float* W_dr   = (const float*)d_in[6];
  const float* b_dr   = (const float*)d_in[7];
  const float* W_e1   = (const float*)d_in[8];
  const float* b_e1   = (const float*)d_in[9];
  const float* W_e2   = (const float*)d_in[10];
  const float* b_e2   = (const float*)d_in[11];
  const float* W_self = (const float*)d_in[12];
  const float* b_self = (const float*)d_in[13];
  const float* W_out  = (const float*)d_in[14];
  const float* b_out  = (const float*)d_in[15];
  const float* ln_g   = (const float*)d_in[16];
  const float* ln_b   = (const float*)d_in[17];
  const float* W_proj = (const float*)d_in[18];
  const float* b_proj = (const float*)d_in[19];
  float* out = (float*)d_out;

  const int N = in_sizes[0] / 16;
  const int E = in_sizes[1] / 2;
  const int B = in_sizes[2] / 51;
  const int L = in_sizes[8] / (256 * 128);
  const int* src = eidx;
  const int* dst = eidx + E;

  // workspace carve (256B aligned slots)
  char* ws = (char*)d_ws;
  size_t off = 0;
  auto carve = [&](size_t bytes) -> void* {
    void* p = ws + off;
    off += (bytes + 255) & ~(size_t)255;
    return p;
  };
  float*  deg     = (float*) carve((size_t)N * 4);
  float*  deg_inv = (float*) carve((size_t)N * 4);
  float*  demb    = (float*) carve((size_t)B * 128 * 4);
  float*  h       = (float*) carve((size_t)N * 128 * 4);
  float*  aggr    = (float*) carve((size_t)N * 128 * 4);
  __bf16* pWe1    = (__bf16*)carve((size_t)L * 256 * 128 * 2);
  __bf16* pWe2    = (__bf16*)carve((size_t)L * 128 * 128 * 2);
  __bf16* pWself  = (__bf16*)carve((size_t)L * 128 * 128 * 2);
  __bf16* pWout   = (__bf16*)carve((size_t)L * 256 * 128 * 2);
  (void)ws_size; (void)n_in; (void)out_size;

  // degree / deg_inv
  zero_f32_k<<<1024, 256, 0, stream>>>(deg, (long)N);
  deg_k<<<(E + 255) / 256, 256, 0, stream>>>(dst, deg, E);
  deginv_k<<<(N + 255) / 256, 256, 0, stream>>>(deg, deg_inv, N);

  // embeddings: h = x@W_node + b_node + (drone@W_drone + b_drone)[batch]
  drone_embed_k<<<(B * 128 + 255) / 256, 256, 0, stream>>>(drone, W_dr, b_dr, demb, B);
  node_embed_k<<<(int)(((long)N * 128 + 255) / 256), 256, 0, stream>>>(x, W_node, b_node, demb, batch, h, N);

  // pack weights into bf16 WMMA fragment order
  for (int l = 0; l < L; ++l) {
    pack_w_k<<<(256 * 128 + 255) / 256, 256, 0, stream>>>(W_e1  + (size_t)l * 256 * 128, pWe1   + (size_t)l * 256 * 128, 256);
    pack_w_k<<<(128 * 128 + 255) / 256, 256, 0, stream>>>(W_e2  + (size_t)l * 128 * 128, pWe2   + (size_t)l * 128 * 128, 128);
    pack_w_k<<<(128 * 128 + 255) / 256, 256, 0, stream>>>(W_self+ (size_t)l * 128 * 128, pWself + (size_t)l * 128 * 128, 128);
    pack_w_k<<<(256 * 128 + 255) / 256, 256, 0, stream>>>(W_out + (size_t)l * 256 * 128, pWout  + (size_t)l * 256 * 128, 256);
  }

  // L EdgeSAGE layers (E and N are multiples of 16 for this harness)
  const int eTiles = E / 16;
  const int nTiles = N / 16;
  for (int l = 0; l < L; ++l) {
    zero_f32_k<<<2048, 256, 0, stream>>>(aggr, (long)N * 128);
    edge_mlp_scatter_k<<<2048, EW * 32, 0, stream>>>(
        h, src, dst,
        pWe1 + (size_t)l * 256 * 128, pWe2 + (size_t)l * 128 * 128,
        b_e1 + (size_t)l * 128, b_e2 + (size_t)l * 128,
        aggr, eTiles);
    node_update_k<<<512, NW * 32, 0, stream>>>(
        h, aggr, deg_inv,
        pWself + (size_t)l * 128 * 128, pWout + (size_t)l * 256 * 128,
        b_self + (size_t)l * 128, b_out + (size_t)l * 128,
        ln_g + (size_t)l * 128, ln_b + (size_t)l * 128,
        nTiles);
  }

  // out = h @ W_proj + b_proj
  proj_k<<<(N + 255) / 256, 256, 0, stream>>>(h, W_proj, b_proj, out, N);
}
